// PerceiverRetriever_51135880626371
// MI455X (gfx1250) — compile-verified, matrix operand order checked by reading, not run on hardware
//
#include <hip/hip_runtime.h>
#include <hip/hip_bf16.h>
#include <stdint.h>

#define HEADS     8
#define DIM_HEAD  64
#define INNER     512
#define QUERY_DIM 512
#define REPS_DIM  512
#define LABELS_DIM 64
#define BUF_DIM   576
#define BB        4
#define NN        2048
#define MM        65536
#define TOPK      16
#define ATT_SCALE 0.125f
#define BK_STRIDE 520   // 512 + 8 bf16 pad -> 1040B row stride, 4-bank rotation

typedef __bf16 bf16_t;
typedef __attribute__((ext_vector_type(16))) __bf16 v16bf;
typedef __attribute__((ext_vector_type(8)))  __bf16 v8bf;
typedef __attribute__((ext_vector_type(8)))  float  v8f;
typedef __attribute__((ext_vector_type(4)))  int    v4i;

// async-to-LDS pointer types: AS1 (global) / AS3 (LDS), 128-bit payload
typedef __attribute__((address_space(1))) v4i* as1_b128_t;
typedef __attribute__((address_space(3))) v4i* as3_b128_t;

// ---------------------------------------------------------------------------
// f32 -> bf16 flat convert
// ---------------------------------------------------------------------------
__global__ __launch_bounds__(256) void cvt_f32_bf16(const float* __restrict__ in,
                                                    bf16_t* __restrict__ out, int n) {
  int i = blockIdx.x * blockDim.x + threadIdx.x;
  int stride = gridDim.x * blockDim.x;
  for (; i < n; i += stride) out[i] = (bf16_t)in[i];
}

// W (K x N, f32) -> Wt (N x K, bf16)
__global__ __launch_bounds__(256) void transpose_cvt(const float* __restrict__ W,
                                                     bf16_t* __restrict__ Wt,
                                                     int K, int N) {
  int e = blockIdx.x * blockDim.x + threadIdx.x;
  if (e >= K * N) return;
  int n = e / K, k = e - n * K;
  Wt[e] = (bf16_t)W[(size_t)k * N + n];
}

// ---------------------------------------------------------------------------
// bf16 WMMA GEMM: C(MxN) = A(MxK) * B(KxN), B given transposed (NxK).
// Block = 8 waves -> 128 rows x 64 cols. B panel (64 x K bf16) staged once
// into LDS via GLOBAL_LOAD_ASYNC_TO_LDS_B128 (ASYNCcnt), then fed to WMMA
// via ds_load_b128. A streams from global with prefetch.
// ---------------------------------------------------------------------------
template <bool OUT_BF16, bool ADD_BIAS>
__global__ __launch_bounds__(256)
void gemm_bf16_wmma(const bf16_t* __restrict__ A, const bf16_t* __restrict__ Bt,
                    float* __restrict__ Cf, bf16_t* __restrict__ Cb,
                    const float* __restrict__ bias, int M, int N, int K) {
  extern __shared__ bf16_t bpanel[];   // 64 rows x BK_STRIDE bf16
  const int tid   = threadIdx.x;
  const int wave  = tid >> 5;
  const int lane  = tid & 31;
  const int m16   = lane & 15;       // row (A) / col (B) within tile
  const int khalf = lane >> 4;       // which K-half this lane holds
  const int rowBase = blockIdx.x * 128 + wave * 16;
  const int colBase = blockIdx.y * 64;

  // ---- stage B panel: 64 rows x K bf16, in 16-byte chunks ----
  const int cpr = K >> 3;            // 8-bf16 chunks per row
  for (int e = tid; e < 64 * cpr; e += 256) {
    int n = e / cpr;
    int c = e - n * cpr;
    const bf16_t* src = Bt + (size_t)(colBase + n) * K + c * 8;
    bf16_t*       dst = bpanel + n * BK_STRIDE + c * 8;
#if __has_builtin(__builtin_amdgcn_global_load_async_to_lds_b128)
    __builtin_amdgcn_global_load_async_to_lds_b128(
        (as1_b128_t)const_cast<bf16_t*>(src), (as3_b128_t)dst, 0, 0);
#else
    *(v8bf*)dst = *(const v8bf*)src;
#endif
  }
#if __has_builtin(__builtin_amdgcn_global_load_async_to_lds_b128)
#if __has_builtin(__builtin_amdgcn_s_wait_asynccnt)
  __builtin_amdgcn_s_wait_asynccnt(0);
#else
  asm volatile("s_wait_asynccnt 0x0" ::: "memory");
#endif
#endif
  __syncthreads();

  v8f acc[4] = {};
  const bf16_t* arow = A + (size_t)(rowBase + m16) * K;

  for (int kk = 0; kk < K; kk += 32) {
    if (kk + 32 < K) __builtin_prefetch(arow + kk + 32, 0, 1);  // global_prefetch_b8
    // A fragment: 16x32 bf16, K-contiguous runs of 8 per lane
    v8bf alo = *(const v8bf*)(arow + kk + khalf * 8);
    v8bf ahi = *(const v8bf*)(arow + kk + 16 + khalf * 8);
    v16bf a;
#pragma unroll
    for (int i = 0; i < 8; ++i) { a[i] = alo[i]; a[i + 8] = ahi[i]; }

#pragma unroll
    for (int t4 = 0; t4 < 4; ++t4) {
      const bf16_t* bp = bpanel + (t4 * 16 + m16) * BK_STRIDE + kk + khalf * 8;
      v8bf blo = *(const v8bf*)bp;          // ds_load_b128
      v8bf bhi = *(const v8bf*)(bp + 16);   // ds_load_b128
      v16bf b;
#pragma unroll
      for (int i = 0; i < 8; ++i) { b[i] = blo[i]; b[i + 8] = bhi[i]; }
      acc[t4] = __builtin_amdgcn_wmma_f32_16x16x32_bf16(
          false, a, false, b, (short)0, acc[t4], false, false);
    }
  }

  // C/D layout: VGPR r -> lanes 0-15: (M=r, N=lane); lanes 16-31: (M=r+8, N=lane-16)
#pragma unroll
  for (int t4 = 0; t4 < 4; ++t4) {
#pragma unroll
    for (int r = 0; r < 8; ++r) {
      int row = rowBase + r + (khalf << 3);
      int col = colBase + t4 * 16 + m16;
      float v = acc[t4][r];
      if (OUT_BF16) {
        Cb[(size_t)row * N + col] = (bf16_t)v;
      } else {
        if (ADD_BIAS) v += bias[col];
        Cf[(size_t)row * N + col] = v;
      }
    }
  }
}

// ---------------------------------------------------------------------------
// Exact f32 path: q0 = x[:,0,:] @ Wq ; e0 = q0 @ Wemb   (one block per batch)
// ---------------------------------------------------------------------------
__global__ __launch_bounds__(256)
void e0_kernel(const float* __restrict__ x, const float* __restrict__ Wq,
               const float* __restrict__ Wemb, float* __restrict__ e0) {
  __shared__ float x0[QUERY_DIM];
  __shared__ float q0[INNER];
  const int b = blockIdx.x, t = threadIdx.x;
  x0[t]       = x[(size_t)b * NN * QUERY_DIM + t];
  x0[t + 256] = x[(size_t)b * NN * QUERY_DIM + t + 256];
  __syncthreads();
  for (int c = t; c < INNER; c += 256) {
    float acc = 0.f;
    for (int k = 0; k < QUERY_DIM; ++k) acc += x0[k] * Wq[(size_t)k * INNER + c];
    q0[c] = acc;
  }
  __syncthreads();
  for (int c = t; c < REPS_DIM; c += 256) {
    float acc = 0.f;
    for (int k = 0; k < INNER; ++k) acc += q0[k] * Wemb[(size_t)k * REPS_DIM + c];
    e0[b * REPS_DIM + c] = acc;
  }
}

// ---------------------------------------------------------------------------
// dist2[b,m] = || context[b,m,:512] - e0[b] ||^2  — one wave per row
// ---------------------------------------------------------------------------
__global__ __launch_bounds__(256)
void dist_kernel(const float* __restrict__ context, const float* __restrict__ e0,
                 float* __restrict__ dist2) {
  const int gw   = (blockIdx.x * blockDim.x + threadIdx.x) >> 5;
  const int lane = threadIdx.x & 31;
  if (gw >= BB * MM) return;
  const int b = gw >> 16;            // MM = 65536
  const int m = gw & (MM - 1);
  const float4* crow = (const float4*)(context + ((size_t)b * MM + m) * BUF_DIM);
  const float4* ev   = (const float4*)(e0 + b * REPS_DIM);
  float acc = 0.f;
#pragma unroll
  for (int i = 0; i < 4; ++i) {
    float4 c = crow[lane + i * 32];
    float4 e = ev[lane + i * 32];
    float dx = c.x - e.x, dy = c.y - e.y, dz = c.z - e.z, dw = c.w - e.w;
    acc += dx * dx + dy * dy + dz * dz + dw * dw;
  }
  for (int off = 16; off; off >>= 1) acc += __shfl_xor(acc, off, 32);
  if (lane == 0) dist2[(size_t)b * MM + m] = acc;
}

// ---------------------------------------------------------------------------
// top-16 smallest per batch: per-thread sorted insertion + LDS pairwise merge
// ---------------------------------------------------------------------------
__global__ __launch_bounds__(256)
void topk_kernel(const float* __restrict__ dist2, int* __restrict__ idx) {
  __shared__ float sval[256 * TOPK];
  __shared__ int   sidx[256 * TOPK];
  const int b = blockIdx.x, t = threadIdx.x;
  float bv[TOPK]; int bi[TOPK];
#pragma unroll
  for (int i = 0; i < TOPK; ++i) { bv[i] = 3.4e38f; bi[i] = 0x7fffffff; }
  for (int m = t; m < MM; m += 256) {
    float v = dist2[(size_t)b * MM + m];
    if (v < bv[TOPK - 1]) {
      int p = TOPK - 1;
      while (p > 0 && v < bv[p - 1]) { bv[p] = bv[p - 1]; bi[p] = bi[p - 1]; --p; }
      bv[p] = v; bi[p] = m;
    }
  }
#pragma unroll
  for (int i = 0; i < TOPK; ++i) { sval[t * TOPK + i] = bv[i]; sidx[t * TOPK + i] = bi[i]; }
  for (int stride = 128; stride >= 1; stride >>= 1) {
    __syncthreads();
    if (t < stride) {
      float av[TOPK], cv[TOPK], rv[TOPK];
      int   ai[TOPK], ci[TOPK], ri[TOPK];
#pragma unroll
      for (int i = 0; i < TOPK; ++i) {
        av[i] = sval[t * TOPK + i];            ai[i] = sidx[t * TOPK + i];
        cv[i] = sval[(t + stride) * TOPK + i]; ci[i] = sidx[(t + stride) * TOPK + i];
      }
      int pa = 0, pc = 0;
#pragma unroll
      for (int i = 0; i < TOPK; ++i) {
        bool takeA = (pc >= TOPK) ||
                     (pa < TOPK && (av[pa] < cv[pc] ||
                                    (av[pa] == cv[pc] && ai[pa] <= ci[pc])));
        if (takeA) { rv[i] = av[pa]; ri[i] = ai[pa]; ++pa; }
        else       { rv[i] = cv[pc]; ri[i] = ci[pc]; ++pc; }
      }
#pragma unroll
      for (int i = 0; i < TOPK; ++i) { sval[t * TOPK + i] = rv[i]; sidx[t * TOPK + i] = ri[i]; }
    }
  }
  __syncthreads();
  if (t < TOPK) idx[b * TOPK + t] = sidx[t];
}

// ---------------------------------------------------------------------------
// gather retrieved rows; k = labels @ Wk ; v = reps @ Wv  (f32, tiny)
// ---------------------------------------------------------------------------
__global__ __launch_bounds__(256)
void gather_kv_kernel(const float* __restrict__ context, const int* __restrict__ idx,
                      const float* __restrict__ Wk, const float* __restrict__ Wv,
                      float* __restrict__ kmat, float* __restrict__ vmat) {
  __shared__ float ret[TOPK * BUF_DIM];   // 36 KB
  __shared__ int   jidx[TOPK];
  const int b = blockIdx.x, t = threadIdx.x;
  if (t < TOPK) jidx[t] = idx[b * TOPK + t];
  __syncthreads();
  for (int e = t; e < TOPK * BUF_DIM; e += 256) {
    int j = e / BUF_DIM, c = e - j * BUF_DIM;
    ret[e] = context[((size_t)b * MM + jidx[j]) * BUF_DIM + c];
  }
  __syncthreads();
  for (int e = t; e < TOPK * INNER; e += 256) {
    int j = e >> 9, c = e & 511;
    float acck = 0.f;
    for (int d = 0; d < LABELS_DIM; ++d)
      acck += ret[j * BUF_DIM + REPS_DIM + d] * Wk[(size_t)d * INNER + c];
    kmat[(size_t)b * TOPK * INNER + e] = acck;
    float accv = 0.f;
    for (int d = 0; d < REPS_DIM; ++d)
      accv += ret[j * BUF_DIM + d] * Wv[(size_t)d * INNER + c];
    vmat[(size_t)b * TOPK * INNER + e] = accv;
  }
}

// ---------------------------------------------------------------------------
// attention: one block per (b,n); one wave per head; 16 keys; bf16 output
// ---------------------------------------------------------------------------
__global__ __launch_bounds__(256)
void attention_kernel(const bf16_t* __restrict__ qbf, const float* __restrict__ kmat,
                      const float* __restrict__ vmat, bf16_t* __restrict__ obf) {
  __shared__ float qs[INNER];
  __shared__ float attnLds[HEADS * TOPK];
  const int row = blockIdx.x;          // b*NN + n
  const int b   = row >> 11;           // NN = 2048
  const int t   = threadIdx.x;
  const bf16_t* qrow = qbf + (size_t)row * INNER;
  qs[t]       = (float)qrow[t];
  qs[t + 256] = (float)qrow[t + 256];
  __syncthreads();

  const int h = t >> 5, lane = t & 31, j = lane & 15, half = lane >> 4;
  const float* krow = kmat + (size_t)b * TOPK * INNER + (size_t)j * INNER + h * DIM_HEAD + half * 32;
  const float* qh   = qs + h * DIM_HEAD + half * 32;
  float s = 0.f;
#pragma unroll
  for (int d = 0; d < 32; ++d) s += qh[d] * krow[d];
  s += __shfl_xor(s, 16, 32);          // both K-halves -> full dot per j
  s *= ATT_SCALE;
  float mx = s;
  for (int off = 8; off; off >>= 1) mx = fmaxf(mx, __shfl_xor(mx, off, 32));
  float e = __expf(s - mx);
  float sum = e;
  for (int off = 8; off; off >>= 1) sum += __shfl_xor(sum, off, 32);
  if (half == 0) attnLds[h * TOPK + j] = e / sum;
  __syncthreads();

  const int c  = t * 2;
  const int h2 = c >> 6;
  float aw[TOPK];
#pragma unroll
  for (int jj = 0; jj < TOPK; ++jj) aw[jj] = attnLds[h2 * TOPK + jj];
  float o0 = 0.f, o1 = 0.f;
  const float* vb = vmat + (size_t)b * TOPK * INNER + c;
#pragma unroll
  for (int jj = 0; jj < TOPK; ++jj) {
    o0 += aw[jj] * vb[jj * INNER];
    o1 += aw[jj] * vb[jj * INNER + 1];
  }
  obf[(size_t)row * INNER + c]     = (bf16_t)o0;
  obf[(size_t)row * INNER + c + 1] = (bf16_t)o1;
}

// ---------------------------------------------------------------------------
extern "C" void kernel_launch(void* const* d_in, const int* in_sizes, int n_in,
                              void* d_out, int out_size, void* d_ws, size_t ws_size,
                              hipStream_t stream) {
  (void)in_sizes; (void)n_in; (void)out_size; (void)ws_size;
  const float* x       = (const float*)d_in[0];
  const float* context = (const float*)d_in[1];
  const float* Wq      = (const float*)d_in[2];
  const float* Wemb    = (const float*)d_in[3];
  const float* Wk      = (const float*)d_in[4];
  const float* Wv      = (const float*)d_in[5];
  const float* Wo      = (const float*)d_in[6];
  const float* bo      = (const float*)d_in[7];
  float* out = (float*)d_out;

  char* ws = (char*)d_ws;
  size_t off = 0;
  auto take = [&](size_t bytes) -> char* {
    char* p = ws + off;
    off = (off + bytes + 255) & ~(size_t)255;
    return p;
  };
  bf16_t* xb    = (bf16_t*)take((size_t)BB * NN * QUERY_DIM * 2);
  bf16_t* Wq_t  = (bf16_t*)take((size_t)QUERY_DIM * INNER * 2);
  bf16_t* Wo_t  = (bf16_t*)take((size_t)INNER * QUERY_DIM * 2);
  bf16_t* q_bf  = (bf16_t*)take((size_t)BB * NN * INNER * 2);
  float*  e0    = (float*) take((size_t)BB * REPS_DIM * 4);
  float*  dist2 = (float*) take((size_t)BB * MM * 4);
  int*    idx   = (int*)   take((size_t)BB * TOPK * 4);
  float*  kmat  = (float*) take((size_t)BB * TOPK * INNER * 4);
  float*  vmat  = (float*) take((size_t)BB * TOPK * INNER * 4);
  bf16_t* attb  = (bf16_t*)take((size_t)BB * NN * INNER * 2);

  cvt_f32_bf16<<<4096, 256, 0, stream>>>(x, xb, BB * NN * QUERY_DIM);
  transpose_cvt<<<(QUERY_DIM * INNER + 255) / 256, 256, 0, stream>>>(Wq, Wq_t, QUERY_DIM, INNER);
  transpose_cvt<<<(INNER * QUERY_DIM + 255) / 256, 256, 0, stream>>>(Wo, Wo_t, INNER, QUERY_DIM);

  const size_t ldsB = (size_t)64 * BK_STRIDE * sizeof(bf16_t);  // 66,560 B
  dim3 gq(BB * NN / 128, INNER / 64);
  gemm_bf16_wmma<true, false><<<gq, 256, ldsB, stream>>>(xb, Wq_t, nullptr, q_bf, nullptr,
                                                         BB * NN, INNER, QUERY_DIM);

  e0_kernel<<<BB, 256, 0, stream>>>(x, Wq, Wemb, e0);
  dist_kernel<<<(BB * MM) / 8, 256, 0, stream>>>(context, e0, dist2);
  topk_kernel<<<BB, 256, 0, stream>>>(dist2, idx);
  gather_kv_kernel<<<BB, 256, 0, stream>>>(context, idx, Wk, Wv, kmat, vmat);
  attention_kernel<<<BB * NN, 256, 0, stream>>>(q_bf, kmat, vmat, attb);

  dim3 go(BB * NN / 128, QUERY_DIM / 64);
  gemm_bf16_wmma<false, true><<<go, 256, ldsB, stream>>>(attb, Wo_t, out, nullptr, bo,
                                                         BB * NN, QUERY_DIM, INNER);
}
